// LocalAstrometryMatcher_36283883717347
// MI455X (gfx1250) — matrix operand order repaired and checked by reading, not compiled, not run on hardware
//
// MI455X (gfx1250) implementation of LocalAstrometryMatcher forward.
//
// Roofline reasoning: the two conv encoders dominate (~0.47 TFLOP). We run them
// as implicit-GEMM bf16 WMMA (v_wmma_f32_16x16x32_bf16, fp32 accum) with NHWC
// bf16 activations (K = Cin*9 contiguous), LDS staging through the gfx1250
// async-to-LDS path (ASYNCcnt), double-buffered across the 9 filter taps.
// GroupNorm statistics are fused into the conv epilogue (each wave's 16-channel
// N-tile == one GN group). Everything else (layer-1 convs with cin<=6, the
// 81-shift correlation, softmax, MLP) is bandwidth-trivial VALU work.
//
// Requires ws_size >= ~404 MB (3 x 134MB bf16 activation buffers + scratch).

#include <hip/hip_runtime.h>
#include <hip/hip_bf16.h>
#include <math.h>

#define HH   128
#define WW   128
#define CCH  64
#define BATCH 64
#define NPIX (HH*WW)

typedef __attribute__((ext_vector_type(16))) __bf16 v16bf;
typedef __attribute__((ext_vector_type(8)))  float  v8f;
typedef __attribute__((ext_vector_type(4)))  unsigned int u32x4;

union Frag { u32x4 q[2]; v16bf v; };   // 32B: 16 bf16 = one WMMA A/B fragment per lane

// ---------- bf16 helpers (raw ushort storage) ----------
__device__ __forceinline__ unsigned short f2bf(float f) {
    union { float f; unsigned u; } v; v.f = f;
    unsigned r = v.u + 0x7FFFu + ((v.u >> 16) & 1u);   // round-to-nearest-even
    return (unsigned short)(r >> 16);
}
__device__ __forceinline__ float bf2f(unsigned short h) {
    union { unsigned u; float f; } v; v.u = ((unsigned)h) << 16; return v.f;
}
__device__ __forceinline__ float bflo(unsigned u) {
    union { unsigned u; float f; } v; v.u = u << 16; return v.f;
}
__device__ __forceinline__ float bfhi(unsigned u) {
    union { unsigned u; float f; } v; v.u = u & 0xFFFF0000u; return v.f;
}
__device__ __forceinline__ float gelu_exact(float x) {
    return 0.5f * x * (1.0f + erff(x * 0.70710678118654752f));
}

// ---------- gfx1250 async global->LDS copy (ASYNCcnt-tracked) ----------
// Generic pointers to __shared__ carry the LDS byte offset in their low 32 bits
// (shared aperture in the high half), which is what the VDST operand of
// global_load_async_to_lds_b128 expects.
__device__ __forceinline__ unsigned lds_off(const void* p) {
    return (unsigned)(unsigned long long)p;
}
__device__ __forceinline__ void async_b128(unsigned ldsoff, const void* gsrc) {
    asm volatile("global_load_async_to_lds_b128 %0, %1, off"
                 :: "v"(ldsoff), "v"((unsigned long long)gsrc) : "memory");
}
template <int N> __device__ __forceinline__ void wait_async() {
    asm volatile("s_wait_asynccnt %0" :: "i"(N) : "memory");
}

// =====================================================================
// Kernel: zero small scratch (stats/logits/pools/coarse)
// =====================================================================
__global__ void zero_kernel(float* p, int n) {
    int i = blockIdx.x * 256 + threadIdx.x;
    if (i < n) p[i] = 0.f;
}

// =====================================================================
// Kernel: weight prep. Conv layers 2-4 (both encoders) -> bf16 slabs laid out
// [tap(ky,kx)][co][ci] so the B tile is a straight [co][ci] LDS copy.
// Also casts the two 64x64 projection matrices to bf16 [o][c].
// =====================================================================
__global__ void prep_kernel(const float* __restrict__ w0, const float* __restrict__ w1,
                            const float* __restrict__ w2, const float* __restrict__ w3,
                            const float* __restrict__ w4, const float* __restrict__ w5,
                            const float* __restrict__ pR, const float* __restrict__ pV,
                            unsigned short* __restrict__ wbuf) {
    int idx = blockIdx.x * 256 + threadIdx.x;
    const int SLAB = 9 * 64 * 64;                       // 36864
    if (idx < 6 * SLAB) {
        int l = idx / SLAB, r = idx % SLAB;
        int pos = r >> 12, co = (r >> 6) & 63, ci = r & 63;
        int ky = pos / 3, kx = pos % 3;
        const float* w = (l == 0) ? w0 : (l == 1) ? w1 : (l == 2) ? w2
                       : (l == 3) ? w3 : (l == 4) ? w4 : w5;
        wbuf[idx] = f2bf(w[((co * 64 + ci) * 3 + ky) * 3 + kx]);
    } else if (idx < 6 * SLAB + 8192) {
        int r = idx - 6 * SLAB;
        wbuf[idx] = f2bf(r < 4096 ? pR[r] : pV[r - 4096]);
    }
}

// =====================================================================
// Kernel: layer-1 direct conv (cin = 6 or 1), NCHW fp32 -> NHWC bf16,
// fused GN stat accumulation. One thread per output pixel.
// =====================================================================
__global__ void conv1_kernel(const float* __restrict__ src, const float* __restrict__ wgt,
                             unsigned short* __restrict__ dst, float* __restrict__ stats,
                             int cin) {
    __shared__ float ws[64 * 6 * 9];
    __shared__ float gred[8];
    const int b = blockIdx.y, t = threadIdx.x;
    const int px = blockIdx.x * 256 + t;
    for (int i = t; i < 64 * cin * 9; i += 256) ws[i] = wgt[i];
    if (t < 8) gred[t] = 0.f;
    __syncthreads();
    const int hh = px >> 7, ww = px & 127;
    float patch[54];
    for (int ci = 0; ci < cin; ++ci)
        for (int ky = 0; ky < 3; ++ky)
            for (int kx = 0; kx < 3; ++kx) {
                int y = hh + ky - 1, x = ww + kx - 1;
                patch[(ci * 3 + ky) * 3 + kx] =
                    ((unsigned)y < 128u && (unsigned)x < 128u)
                        ? src[(((size_t)b * cin + ci) * HH + y) * WW + x] : 0.f;
            }
    float gs[4] = {0, 0, 0, 0}, gss[4] = {0, 0, 0, 0};
    const int kn = cin * 9;
    for (int co = 0; co < 64; ++co) {
        float a = 0.f;
        const float* wr = &ws[co * kn];
        for (int k = 0; k < kn; ++k) a = fmaf(patch[k], wr[k], a);
        dst[((size_t)b * NPIX + px) * 64 + co] = f2bf(a);
        gs[co >> 4] += a; gss[co >> 4] += a * a;
    }
    for (int g = 0; g < 4; ++g) {
        atomicAdd(&gred[g * 2], gs[g]);
        atomicAdd(&gred[g * 2 + 1], gss[g]);
    }
    __syncthreads();
    if (t < 8) atomicAdd(&stats[b * 8 + t], gred[t]);
}

// =====================================================================
// Kernel: WMMA implicit-GEMM 3x3 conv, 64->64 ch.
// Block = 128 thr (4 waves), tile = 8x8 pixels (M=64) x 64 out-ch (N=64).
// K = 9 taps x 64 ci, staged through LDS with async-to-LDS, double buffered.
// Wave w owns N-tile w (= GN group w): 4 M-tiles x 18 K-steps = 72 WMMAs.
// =====================================================================
__global__ __launch_bounds__(128) void conv_wmma_kernel(
        const unsigned short* __restrict__ src,    // [B][H][W][64] bf16
        const unsigned short* __restrict__ wslab,  // [9][co][ci] bf16
        unsigned short* __restrict__ dst,          // [B][H][W][64] bf16
        float* __restrict__ stats) {               // [B][4 groups][2]
    __shared__ __align__(16) unsigned short As[2][64 * 64];
    __shared__ __align__(16) unsigned short Bs[2][64 * 64];
    const int b = blockIdx.y;
    const int ty = blockIdx.x >> 4, tx = blockIdx.x & 15;   // 16x16 tiles of 8x8 px
    const int t = threadIdx.x, lane = t & 31, wid = t >> 5;

    auto issue = [&](int pos, int buf) {
        const int ky = pos / 3, kx = pos % 3;
        const int p = t >> 1, halfc = (t & 1) * 32;
        // A tile: 64 shifted pixels x 64 ci (zero-fill SAME padding)
        {
            const int hh = ty * 8 + (p >> 3) + ky - 1;
            const int wv = tx * 8 + (p & 7) + kx - 1;
            unsigned la = lds_off(&As[buf][p * 64 + halfc]);
            if ((unsigned)hh < 128u && (unsigned)wv < 128u) {
                const unsigned short* g = src + (((size_t)b * HH + hh) * WW + wv) * 64 + halfc;
                for (int j = 0; j < 4; ++j) async_b128(la + j * 16, g + j * 8);
            } else {
                u32x4 z = {0u, 0u, 0u, 0u};
                u32x4* q = (u32x4*)&As[buf][p * 64 + halfc];
                for (int j = 0; j < 4; ++j) q[j] = z;
            }
        }
        // B tile: [co][ci] straight copy of this tap's weight slab
        {
            const int co = t >> 1;
            const unsigned short* g = wslab + ((size_t)pos * 64 + co) * 64 + halfc;
            unsigned lb = lds_off(&Bs[buf][co * 64 + halfc]);
            for (int j = 0; j < 4; ++j) async_b128(lb + j * 16, g + j * 8);
        }
    };

    v8f z8 = {0.f, 0.f, 0.f, 0.f, 0.f, 0.f, 0.f, 0.f};
    v8f acc[4] = {z8, z8, z8, z8};

    auto compute = [&](int buf) {
        const int half = lane >> 4, n = lane & 15;
        const int co = wid * 16 + n;
        for (int kk = 0; kk < 2; ++kk) {
            // B 32x16 layout: lanes 0-15 hold K 0-15, lanes 16-31 hold K 16-31 (contig)
            const int kbB = kk * 32 + half * 16;
            Frag bf;
            bf.q[0] = *(const u32x4*)&Bs[buf][co * 64 + kbB];
            bf.q[1] = *(const u32x4*)&Bs[buf][co * 64 + kbB + 8];
            // A 16x32 layout: lanes 0-15 hold K 0-15, lanes 16-31 hold K 16-31 (contig)
            const int kbA = kk * 32 + half * 16;
            for (int mt = 0; mt < 4; ++mt) {
                const int px = mt * 16 + n;
                Frag af;
                af.q[0] = *(const u32x4*)&As[buf][px * 64 + kbA];
                af.q[1] = *(const u32x4*)&As[buf][px * 64 + kbA + 8];
                acc[mt] = __builtin_amdgcn_wmma_f32_16x16x32_bf16(
                    false, af.v, false, bf.v, (short)0, acc[mt], false, false);
            }
        }
    };

    issue(0, 0);
    for (int pos = 0; pos < 9; ++pos) {
        if (pos < 8) { issue(pos + 1, (pos + 1) & 1); wait_async<8>(); }
        else         { wait_async<0>(); }
        __syncthreads();
        compute(pos & 1);
        __syncthreads();
    }

    // Epilogue: store bf16 NHWC + fused GN sum/sumsq (wave's N-tile == GN group wid)
    float s = 0.f, ss = 0.f;
    const int n = lane & 15, half = lane >> 4, co = wid * 16 + n;
    for (int mt = 0; mt < 4; ++mt)
        for (int r = 0; r < 8; ++r) {
            const int px = mt * 16 + r + half * 8;
            const int hh = ty * 8 + (px >> 3), wv = tx * 8 + (px & 7);
            float v = acc[mt][r];
            dst[(((size_t)b * HH + hh) * WW + wv) * 64 + co] = f2bf(v);
            s += v; ss += v * v;
        }
    for (int off = 16; off; off >>= 1) {
        s += __shfl_down(s, off, 32);
        ss += __shfl_down(ss, off, 32);
    }
    if (lane == 0) {
        atomicAdd(&stats[b * 8 + wid * 2], s);
        atomicAdd(&stats[b * 8 + wid * 2 + 1], ss);
    }
}

// =====================================================================
// Kernel: GroupNorm + exact GELU, in place on bf16 NHWC activations.
// =====================================================================
__global__ void gn_gelu_kernel(unsigned short* __restrict__ act,
                               const float* __restrict__ stats,
                               const float* __restrict__ gamma,
                               const float* __restrict__ beta) {
    size_t idx = (size_t)blockIdx.x * 256 + threadIdx.x;   // B*NPIX*64 elements
    const int c = (int)(idx & 63);
    const int b = (int)(idx >> 20);                        // /(NPIX*64)
    const int g = c >> 4;
    const float cnt = 16.f * (float)NPIX;
    float S = stats[b * 8 + g * 2], SS = stats[b * 8 + g * 2 + 1];
    float m = S / cnt;
    float var = SS / cnt - m * m;
    float r = rsqrtf(var + 1e-5f);
    float x = bf2f(act[idx]);
    float y = (x - m) * r * gamma[c] + beta[c];
    act[idx] = f2bf(gelu_exact(y));
}

// =====================================================================
// Kernel: spatial mean pool over H,W per (b,c).
// =====================================================================
__global__ void pool_kernel(const unsigned short* __restrict__ feat, float* __restrict__ pool) {
    const int b = blockIdx.x, t = threadIdx.x;
    const int c = t & 63, part = t >> 6;
    float acc = 0.f;
    for (int px = part; px < NPIX; px += 4)
        acc += bf2f(feat[((size_t)b * NPIX + px) * 64 + c]);
    __shared__ float red[256];
    red[t] = acc;
    __syncthreads();
    if (part == 0)
        pool[b * 64 + c] = (red[c] + red[64 + c] + red[128 + c] + red[192 + c]) * (1.f / NPIX);
}

// =====================================================================
// Kernel: 64x64 channel projection via WMMA (K=64, two K-steps), bf16 out.
// =====================================================================
__global__ __launch_bounds__(128) void proj_wmma_kernel(
        const unsigned short* __restrict__ feat,  // [B][NPIX][64]
        const unsigned short* __restrict__ pw,    // [o][c] bf16
        unsigned short* __restrict__ dst) {
    __shared__ __align__(16) unsigned short As[64 * 64];
    __shared__ __align__(16) unsigned short Bs[64 * 64];
    const int b = blockIdx.y, p0 = blockIdx.x * 64;
    const int t = threadIdx.x, lane = t & 31, wid = t >> 5;
    {
        const int p = t >> 1, halfc = (t & 1) * 32;
        const unsigned short* gA = feat + ((size_t)b * NPIX + p0 + p) * 64 + halfc;
        unsigned la = lds_off(&As[p * 64 + halfc]);
        for (int j = 0; j < 4; ++j) async_b128(la + j * 16, gA + j * 8);
        const unsigned short* gB = pw + (size_t)p * 64 + halfc;
        unsigned lb = lds_off(&Bs[p * 64 + halfc]);
        for (int j = 0; j < 4; ++j) async_b128(lb + j * 16, gB + j * 8);
    }
    wait_async<0>();
    __syncthreads();
    v8f z8 = {0.f, 0.f, 0.f, 0.f, 0.f, 0.f, 0.f, 0.f};
    v8f acc[4] = {z8, z8, z8, z8};
    const int half = lane >> 4, n = lane & 15, co = wid * 16 + n;
    for (int kk = 0; kk < 2; ++kk) {
        const int kbB = kk * 32 + half * 16;
        Frag bf;
        bf.q[0] = *(const u32x4*)&Bs[co * 64 + kbB];
        bf.q[1] = *(const u32x4*)&Bs[co * 64 + kbB + 8];
        const int kbA = kk * 32 + half * 16;
        for (int mt = 0; mt < 4; ++mt) {
            const int px = mt * 16 + n;
            Frag af;
            af.q[0] = *(const u32x4*)&As[px * 64 + kbA];
            af.q[1] = *(const u32x4*)&As[px * 64 + kbA + 8];
            acc[mt] = __builtin_amdgcn_wmma_f32_16x16x32_bf16(
                false, af.v, false, bf.v, (short)0, acc[mt], false, false);
        }
    }
    for (int mt = 0; mt < 4; ++mt)
        for (int r = 0; r < 8; ++r) {
            const int px = mt * 16 + r + half * 8;
            dst[((size_t)b * NPIX + p0 + px) * 64 + co] = f2bf(acc[mt][r]);
        }
}

// =====================================================================
// Kernel: per-pixel L2 normalization over channels (in place).
// =====================================================================
__global__ void l2norm_kernel(unsigned short* __restrict__ nmap) {
    size_t px = (size_t)blockIdx.x * 256 + threadIdx.x;    // B*NPIX
    unsigned* p = (unsigned*)(nmap + px * 64);
    unsigned v[32];
    float s = 0.f;
    for (int i = 0; i < 32; ++i) {
        v[i] = p[i];
        float a = bflo(v[i]), c = bfhi(v[i]);
        s += a * a + c * c;
    }
    float inv = 1.f / fmaxf(sqrtf(s), 1e-6f);
    for (int i = 0; i < 32; ++i) {
        unsigned lo = f2bf(bflo(v[i]) * inv);
        unsigned hi = f2bf(bfhi(v[i]) * inv);
        p[i] = lo | (hi << 16);
    }
}

// =====================================================================
// Kernel: 81-shift (±4) cross-correlation with edge padding.
// Block = 128 thr, tile = 16x8 pixels; vis halo (24x16) staged in LDS.
// =====================================================================
__global__ __launch_bounds__(128) void corr_kernel(
        const unsigned short* __restrict__ rn, const unsigned short* __restrict__ vn,
        float* __restrict__ logits) {
    __shared__ __align__(16) unsigned short Vt[24 * 16 * 64];   // 48KB
    __shared__ float wred[4];
    const int b = blockIdx.y;
    const int tyy = blockIdx.x >> 4, txx = blockIdx.x & 15;     // 8x16 grid of 16x8 tiles
    const int gy0 = tyy * 16, gx0 = txx * 8;
    const int t = threadIdx.x, lane = t & 31, wid = t >> 5;
    for (int i = t; i < 24 * 16; i += 128) {
        const int lh = i >> 4, lw = i & 15;
        const int gh = min(127, max(0, gy0 + lh - 4));
        const int gw = min(127, max(0, gx0 + lw - 4));
        const u32x4* g = (const u32x4*)(vn + (((size_t)b * HH + gh) * WW + gw) * 64);
        u32x4* d = (u32x4*)(Vt + i * 64);
        for (int j = 0; j < 8; ++j) d[j] = g[j];
    }
    const int lh = t >> 3, lw = t & 7;
    unsigned r[32];
    {
        const unsigned* g = (const unsigned*)(rn + (((size_t)b * HH + gy0 + lh) * WW + gx0 + lw) * 64);
        for (int i = 0; i < 32; ++i) r[i] = g[i];
    }
    __syncthreads();
    for (int s = 0; s < 81; ++s) {
        const int dy = s / 9 - 4, dx = s % 9 - 4;
        const unsigned* v = (const unsigned*)(Vt + (((lh + dy + 4) << 4) + (lw + dx + 4)) * 64);
        float acc = 0.f;
        for (int i = 0; i < 32; ++i) {
            unsigned rv = r[i], vv = v[i];
            acc = fmaf(bflo(rv), bflo(vv), acc);
            acc = fmaf(bfhi(rv), bfhi(vv), acc);
        }
        for (int off = 16; off; off >>= 1) acc += __shfl_down(acc, off, 32);
        if (lane == 0) wred[wid] = acc;
        __syncthreads();
        if (t == 0)
            atomicAdd(&logits[b * 81 + s], wred[0] + wred[1] + wred[2] + wred[3]);
        __syncthreads();
    }
}

// =====================================================================
// Kernel: softmax over 81 shifts, coarse dx/dy, confidence, temperature.
// =====================================================================
__global__ void softmax_kernel(const float* __restrict__ logits_raw,
                               const float* __restrict__ log_temp,
                               float* __restrict__ out, float* __restrict__ coarse) {
    const int b = blockIdx.x, t = threadIdx.x;
    __shared__ float l[81];
    if (t < 81) l[t] = logits_raw[b * 81 + t] * (0.125f / (float)NPIX);
    __syncthreads();
    if (t == 0) {
        const float temp = fmaxf(expf(log_temp[0]), 1e-3f);
        float mx = -1e30f;
        for (int s = 0; s < 81; ++s) mx = fmaxf(mx, l[s] / temp);
        float pr[81], sum = 0.f;
        for (int s = 0; s < 81; ++s) { pr[s] = expf(l[s] / temp - mx); sum += pr[s]; }
        float cdx = 0.f, cdy = 0.f, cmax = 0.f;
        for (int s = 0; s < 81; ++s) {
            const float p = pr[s] / sum;
            cdx += p * (float)(s % 9 - 4);
            cdy += p * (float)(s / 9 - 4);
            cmax = fmaxf(cmax, p);
        }
        out[128 + b] = cdx; out[192 + b] = cdy; out[448 + b] = cmax;
        coarse[b * 2] = cdx; coarse[b * 2 + 1] = cdy;
        if (b == 0) out[512] = temp;
    }
}

// =====================================================================
// Kernel: MLP head + final outputs (dx_px, dy_px, pred_sky, log_sigma).
// =====================================================================
__global__ __launch_bounds__(128) void mlp_kernel(
        const float* __restrict__ poolR, const float* __restrict__ poolV,
        const float* __restrict__ coarse,
        const float* __restrict__ w1, const float* __restrict__ b1,
        const float* __restrict__ w2, const float* __restrict__ b2,
        const float* __restrict__ w3, const float* __restrict__ b3,
        const float* __restrict__ p2s, float* __restrict__ out) {
    const int b = blockIdx.x, t = threadIdx.x;
    __shared__ float in[194], h1[128], h2[128];
    if (t < 64) {
        float r = poolR[b * 64 + t], v = poolV[b * 64 + t];
        in[t] = r; in[64 + t] = v; in[128 + t] = r - v;
    }
    if (t == 0) { in[192] = coarse[b * 2]; in[193] = coarse[b * 2 + 1]; }
    __syncthreads();
    float a = b1[t];
    for (int k = 0; k < 194; ++k) a = fmaf(in[k], w1[t * 194 + k], a);
    h1[t] = gelu_exact(a);
    __syncthreads();
    a = b2[t];
    for (int k = 0; k < 128; ++k) a = fmaf(h1[k], w2[t * 128 + k], a);
    h2[t] = gelu_exact(a);
    __syncthreads();
    if (t == 0) {
        float r0 = b3[0], r1 = b3[1], r2 = b3[2];
        for (int k = 0; k < 128; ++k) {
            r0 = fmaf(h2[k], w3[k], r0);
            r1 = fmaf(h2[k], w3[128 + k], r1);
            r2 = fmaf(h2[k], w3[256 + k], r2);
        }
        const float cdx = coarse[b * 2], cdy = coarse[b * 2 + 1];
        const float dx = cdx + r0, dy = cdy + r1;
        const float ls = fminf(fmaxf(r2, -6.f), 3.f);
        const float* M = p2s + b * 4;
        out[b] = dx;
        out[64 + b] = dy;
        out[256 + b * 2]     = M[0] * dx + M[1] * dy;
        out[256 + b * 2 + 1] = M[2] * dx + M[3] * dy;
        out[384 + b] = ls;
    }
}

// =====================================================================
// Host launcher
// =====================================================================
extern "C" void kernel_launch(void* const* d_in, const int* in_sizes, int n_in,
                              void* d_out, int out_size, void* d_ws, size_t ws_size,
                              hipStream_t stream) {
    (void)in_sizes; (void)n_in; (void)out_size; (void)ws_size;
    const float* rubin = (const float*)d_in[0];
    const float* vis   = (const float*)d_in[1];
    const float* p2s   = (const float*)d_in[2];
    const float *rw[4], *rg[4], *rb[4], *vw[4], *vg[4], *vb[4];
    for (int i = 0; i < 4; ++i) {
        rw[i] = (const float*)d_in[3 + 3 * i];
        rg[i] = (const float*)d_in[4 + 3 * i];
        rb[i] = (const float*)d_in[5 + 3 * i];
        vw[i] = (const float*)d_in[15 + 3 * i];
        vg[i] = (const float*)d_in[16 + 3 * i];
        vb[i] = (const float*)d_in[17 + 3 * i];
    }
    const float* projR = (const float*)d_in[27];
    const float* projV = (const float*)d_in[28];
    const float* logt  = (const float*)d_in[29];
    const float* mw1 = (const float*)d_in[30]; const float* mb1 = (const float*)d_in[31];
    const float* mw2 = (const float*)d_in[32]; const float* mb2 = (const float*)d_in[33];
    const float* mw3 = (const float*)d_in[34]; const float* mb3 = (const float*)d_in[35];
    float* out = (float*)d_out;

    char* ws = (char*)d_ws;
    const size_t ACT = (size_t)BATCH * NPIX * 64 * 2;      // 134,217,728 B
    unsigned short* actA = (unsigned short*)ws;
    unsigned short* actB = (unsigned short*)(ws + ACT);
    unsigned short* nR   = (unsigned short*)(ws + 2 * ACT);
    unsigned short* wbuf = (unsigned short*)(ws + 3 * ACT);           // 458,752 B
    float* smallf = (float*)(ws + 3 * ACT + 512 * 1024);
    float* stats  = smallf;                 // 8 slots * 512
    float* logits = smallf + 4096;          // 64*81
    float* poolR  = smallf + 4096 + 5184;
    float* poolV  = poolR + 4096;
    float* coarse = poolV + 4096;           // 128
    const int SMALLN = 4096 + 5184 + 4096 + 4096 + 128;

    unsigned short* slab[6];
    for (int i = 0; i < 6; ++i) slab[i] = wbuf + i * 36864;
    unsigned short* projRbf = wbuf + 6 * 36864;
    unsigned short* projVbf = projRbf + 4096;

    zero_kernel<<<(SMALLN + 255) / 256, 256, 0, stream>>>(smallf, SMALLN);
    prep_kernel<<<(6 * 36864 + 8192 + 255) / 256, 256, 0, stream>>>(
        rw[1], rw[2], rw[3], vw[1], vw[2], vw[3], projR, projV, wbuf);

    const dim3 gConv(256, 64), gGn((size_t)BATCH * NPIX * 64 / 256);

    // ---- rubin encoder ----
    conv1_kernel<<<dim3(64, 64), 256, 0, stream>>>(rubin, rw[0], actA, stats + 0 * 512, 6);
    gn_gelu_kernel<<<gGn, 256, 0, stream>>>(actA, stats + 0 * 512, rg[0], rb[0]);
    conv_wmma_kernel<<<gConv, 128, 0, stream>>>(actA, slab[0], actB, stats + 1 * 512);
    gn_gelu_kernel<<<gGn, 256, 0, stream>>>(actB, stats + 1 * 512, rg[1], rb[1]);
    conv_wmma_kernel<<<gConv, 128, 0, stream>>>(actB, slab[1], actA, stats + 2 * 512);
    gn_gelu_kernel<<<gGn, 256, 0, stream>>>(actA, stats + 2 * 512, rg[2], rb[2]);
    conv_wmma_kernel<<<gConv, 128, 0, stream>>>(actA, slab[2], actB, stats + 3 * 512);
    gn_gelu_kernel<<<gGn, 256, 0, stream>>>(actB, stats + 3 * 512, rg[3], rb[3]);
    pool_kernel<<<64, 256, 0, stream>>>(actB, poolR);
    proj_wmma_kernel<<<gConv, 128, 0, stream>>>(actB, projRbf, nR);
    l2norm_kernel<<<(BATCH * NPIX) / 256, 256, 0, stream>>>(nR);

    // ---- vis encoder (reuses actA/actB; nV lands in actA) ----
    conv1_kernel<<<dim3(64, 64), 256, 0, stream>>>(vis, vw[0], actA, stats + 4 * 512, 1);
    gn_gelu_kernel<<<gGn, 256, 0, stream>>>(actA, stats + 4 * 512, vg[0], vb[0]);
    conv_wmma_kernel<<<gConv, 128, 0, stream>>>(actA, slab[3], actB, stats + 5 * 512);
    gn_gelu_kernel<<<gGn, 256, 0, stream>>>(actB, stats + 5 * 512, vg[1], vb[1]);
    conv_wmma_kernel<<<gConv, 128, 0, stream>>>(actB, slab[4], actA, stats + 6 * 512);
    gn_gelu_kernel<<<gGn, 256, 0, stream>>>(actA, stats + 6 * 512, vg[2], vb[2]);
    conv_wmma_kernel<<<gConv, 128, 0, stream>>>(actA, slab[5], actB, stats + 7 * 512);
    gn_gelu_kernel<<<gGn, 256, 0, stream>>>(actB, stats + 7 * 512, vg[3], vb[3]);
    pool_kernel<<<64, 256, 0, stream>>>(actB, poolV);
    proj_wmma_kernel<<<gConv, 128, 0, stream>>>(actB, projVbf, actA);
    l2norm_kernel<<<(BATCH * NPIX) / 256, 256, 0, stream>>>(actA);

    // ---- correlation + heads ----
    corr_kernel<<<dim3(128, 64), 128, 0, stream>>>(nR, actA, logits);
    softmax_kernel<<<64, 128, 0, stream>>>(logits, logt, out, coarse);
    mlp_kernel<<<64, 128, 0, stream>>>(poolR, poolV, coarse,
                                       mw1, mb1, mw2, mb2, mw3, mb3, p2s, out);
}
